// PGNNMask_67276367724949
// MI455X (gfx1250) — compile-verified, matrix operand order checked by reading, not run on hardware
//
#include <hip/hip_runtime.h>
#include <hip/hip_bf16.h>
#include <math.h>

#define N_ROWS 8192
#define HDIM 64
#define ZDIM 128
#define NB (N_ROWS / 16) // 512 tile-blocks per dim

typedef __attribute__((ext_vector_type(16))) __bf16 v16bf;
typedef __attribute__((ext_vector_type(8)))  float  v8f;
typedef __attribute__((ext_vector_type(4)))  unsigned int u32x4;

__device__ __forceinline__ unsigned short f2bf(float f) {
  unsigned int u = __float_as_uint(f);
  u += 0x7FFFu + ((u >> 16) & 1u);           // round-to-nearest-even
  return (unsigned short)(u >> 16);
}

// ---------------- K1: Z = relu(h@W1+b1)@W2+b2, stored as bf16 ----------------
__global__ __launch_bounds__(256) void mlp_kernel(const float* __restrict__ h,
                                                  const float* __restrict__ W1,
                                                  const float* __restrict__ b1,
                                                  const float* __restrict__ W2,
                                                  const float* __restrict__ b2,
                                                  unsigned short* __restrict__ Zbf) {
  __shared__ float W1s[HDIM * HDIM];
  __shared__ float W2s[HDIM * ZDIM];
  __shared__ float b1s[HDIM];
  __shared__ float b2s[ZDIM];
  const int t = threadIdx.x;
  for (int i = t; i < HDIM * HDIM; i += 256) W1s[i] = W1[i];
  for (int i = t; i < HDIM * ZDIM; i += 256) W2s[i] = W2[i];
  if (t < HDIM) b1s[t] = b1[t];
  if (t < ZDIM) b2s[t] = b2[t];
  __syncthreads();

  const int row = blockIdx.x * 256 + t;
  const float* hr = h + (size_t)row * HDIM;

  float acc[HDIM];
#pragma unroll
  for (int j = 0; j < HDIM; ++j) acc[j] = b1s[j];
  for (int k = 0; k < HDIM; ++k) {
    const float hk = hr[k];
#pragma unroll
    for (int j = 0; j < HDIM; ++j) acc[j] = fmaf(hk, W1s[k * HDIM + j], acc[j]);
  }
#pragma unroll
  for (int j = 0; j < HDIM; ++j) acc[j] = fmaxf(acc[j], 0.0f);

  unsigned int* zrow = (unsigned int*)(Zbf + (size_t)row * ZDIM);
  for (int o = 0; o < ZDIM; o += 2) {
    float z0 = b2s[o], z1 = b2s[o + 1];
#pragma unroll
    for (int j = 0; j < HDIM; ++j) {
      z0 = fmaf(acc[j], W2s[j * ZDIM + o],     z0);
      z1 = fmaf(acc[j], W2s[j * ZDIM + o + 1], z1);
    }
    zrow[o >> 1] = (unsigned int)f2bf(z0) | ((unsigned int)f2bf(z1) << 16);
  }
}

// --------- K2: upper-tile WMMA GEMM + gumbel-sigmoid + symmetrize + rowsums ---------
__global__ __launch_bounds__(128) void gemm_gate_kernel(const unsigned short* __restrict__ Zbf,
                                                        const float* __restrict__ noise,
                                                        float* __restrict__ outA,   // unnormalized adj (adj_norm slot)
                                                        float* __restrict__ outL,   // adj_logits
                                                        float* __restrict__ rowsum) {
  const int bi = blockIdx.x;
  const int bj = blockIdx.y * 4 + threadIdx.y;   // wave-uniform
  if (bj < bi) return;                           // strict lower tiles skipped

  const int lane = threadIdx.x;                  // wave32 lane
  const int lr   = lane & 15;
  const int sel  = lane >> 4;

  const int gi0 = bi * 16, gj0 = bj * 16;
  const unsigned short* Zi = Zbf + (size_t)(gi0 + lr) * ZDIM;
  const unsigned short* Zj = Zbf + (size_t)(gj0 + lr) * ZDIM;

  // C[m][n] = sum_k Z[gi0+m][k] * Z[gj0+n][k]  (A = Z rows, B = Z^T == Z rows)
  v8f c = {};
#pragma unroll
  for (int kk = 0; kk < ZDIM; kk += 32) {
    const int base = kk + sel * 8;               // per-lane K chunk (16B aligned)
    union { u32x4 u[2]; v16bf v; } fa, fb;
    fa.u[0] = *(const u32x4*)(Zi + base);
    fa.u[1] = *(const u32x4*)(Zi + base + 16);
    fb.u[0] = *(const u32x4*)(Zj + base);
    fb.u[1] = *(const u32x4*)(Zj + base + 16);
    c = __builtin_amdgcn_wmma_f32_16x16x32_bf16(false, fa.v, false, fb.v,
                                                (short)0, c, false, false);
  }

  const size_t Nl = (size_t)N_ROWS;
  const int    gn = gj0 + lr;                    // this lane's column
  const bool   diag = (bi == bj);
  float aval[8];

#pragma unroll
  for (int r = 0; r < 8; ++r) {
    const int gm = gi0 + 8 * sel + r;            // this value's row
    const float L = c[r];
    outL[(size_t)gm * Nl + gn] = L;              // logits (straight tile)
    float a;
    if (diag && gm == gn) {
      a = 1.0f;                                  // fill_diagonal_(1)
    } else {
      const size_t uidx = (gm < gn) ? ((size_t)gm * Nl + gn)
                                    : ((size_t)gn * Nl + gm);  // noise from upper tri
      const float u   = noise[uidx];
      const float eps = (2.0f * 0.0001f - 1.0f) * u + (1.0f - 0.0001f);
      const float g   = __logf(eps) - __logf(1.0f - eps) + L;  // TEMPERATURE == 1
      a = 1.0f / (1.0f + __expf(-g));
    }
    aval[r] = a;
    outA[(size_t)gm * Nl + gn] = a;              // unnormalized adj (straight tile)
  }

  // row sums for rows gi0+m : reduce over the 16 lanes of each half-wave
#pragma unroll
  for (int r = 0; r < 8; ++r) {
    float v = aval[r];
    v += __shfl_xor(v, 1, 32);
    v += __shfl_xor(v, 2, 32);
    v += __shfl_xor(v, 4, 32);
    v += __shfl_xor(v, 8, 32);
    if (lr == 0) atomicAdd(&rowsum[gi0 + 8 * sel + r], v);
  }

  if (!diag) {
    // column sums -> rows gj0+n (symmetric counterpart)
    float cs = 0.0f;
#pragma unroll
    for (int r = 0; r < 8; ++r) cs += aval[r];
    cs += __shfl_xor(cs, 16, 32);
    if (sel == 0) atomicAdd(&rowsum[gn], cs);

    // transposed stores: row gn, 8 consecutive cols gi0+8*sel .. +7 (2x float4)
    const size_t tb = (size_t)gn * Nl + gi0 + 8 * sel;
    *(float4*)(outA + tb)     = make_float4(aval[0], aval[1], aval[2], aval[3]);
    *(float4*)(outA + tb + 4) = make_float4(aval[4], aval[5], aval[6], aval[7]);
    *(float4*)(outL + tb)     = make_float4(c[0], c[1], c[2], c[3]);
    *(float4*)(outL + tb + 4) = make_float4(c[4], c[5], c[6], c[7]);
  }
}

// ---------------- K3: d = rsqrt(rowsum) ----------------
__global__ void dvec_kernel(const float* __restrict__ rowsum, float* __restrict__ dvec) {
  const int i = blockIdx.x * 256 + threadIdx.x;
  if (i < N_ROWS) dvec[i] = rsqrtf(rowsum[i]);
}

// ---------------- K4: in-place adj_norm[i][j] = d[i]*A[i][j]*d[j] ----------------
__global__ __launch_bounds__(256) void norm_kernel(float* __restrict__ A,
                                                   const float* __restrict__ dvec) {
  const size_t e4  = (size_t)blockIdx.x * 256 + threadIdx.x;  // one float4 per thread
  const size_t row = e4 >> 11;                                // 2048 float4 per row
  const size_t c4  = e4 & 2047;
  float4 v = ((float4*)A)[e4];
  const float  di = dvec[row];
  const float4 dj = ((const float4*)dvec)[c4];
  v.x *= di * dj.x; v.y *= di * dj.y; v.z *= di * dj.z; v.w *= di * dj.w;
  ((float4*)A)[e4] = v;
}

extern "C" void kernel_launch(void* const* d_in, const int* in_sizes, int n_in,
                              void* d_out, int out_size, void* d_ws, size_t ws_size,
                              hipStream_t stream) {
  const float* h     = (const float*)d_in[0];
  const float* W1    = (const float*)d_in[1];
  const float* b1    = (const float*)d_in[2];
  const float* W2    = (const float*)d_in[3];
  const float* b2    = (const float*)d_in[4];
  const float* noise = (const float*)d_in[5];

  float* outA = (float*)d_out;                               // adj_norm (staged unnormalized)
  float* outL = outA + (size_t)N_ROWS * N_ROWS;              // adj_logits

  unsigned short* Zbf = (unsigned short*)d_ws;               // 8192*128 bf16 = 2 MB
  float* rowsum = (float*)((char*)d_ws + (size_t)N_ROWS * ZDIM * sizeof(unsigned short));
  float* dvec   = rowsum + N_ROWS;

  hipMemsetAsync(rowsum, 0, N_ROWS * sizeof(float), stream);

  mlp_kernel<<<N_ROWS / 256, 256, 0, stream>>>(h, W1, b1, W2, b2, Zbf);

  dim3 g2(NB, NB / 4), blk2(32, 4);                          // 4 waves/block, 1 tile/wave
  gemm_gate_kernel<<<g2, blk2, 0, stream>>>(Zbf, noise, outA, outL, rowsum);

  dvec_kernel<<<N_ROWS / 256, 256, 0, stream>>>(rowsum, dvec);

  norm_kernel<<<(unsigned)(((size_t)N_ROWS * N_ROWS / 4) / 256), 256, 0, stream>>>(outA, dvec);
}
